// LSTM_27582279975612
// MI455X (gfx1250) — matrix operand order repaired and checked by reading, not compiled, not run on hardware
//
#include <hip/hip_runtime.h>
#include <hip/hip_bf16.h>

// ---------------------------------------------------------------------------
// 3-layer LSTM (SEQ=20, B=256, IN=512, HID=1024, OUT=256) for gfx1250.
// bf16 WMMA (f32 accumulate) + async global->LDS double-buffered staging.
// ---------------------------------------------------------------------------

#define SEQ_LEN 20
#define BATCH   256
#define IN_DIM  512
#define HID     1024
#define OUTD    256

typedef __attribute__((ext_vector_type(8)))  __bf16 v8bf;
typedef __attribute__((ext_vector_type(16))) __bf16 v16bf;
typedef __attribute__((ext_vector_type(8)))  float  v8f;

// --- CDNA5 async copy + waits ----------------------------------------------
__device__ __forceinline__ void async_copy_b128(void* lds_dst, const void* gsrc) {
    // VDST = LDS byte address (low 32 bits of the flat LDS-aperture address),
    // VADDR = 64-bit global address, no SADDR.  Tracked by ASYNCcnt.
    asm volatile("global_load_async_to_lds_b128 %0, %1, off"
                 :: "v"((unsigned int)(unsigned long long)lds_dst), "v"(gsrc)
                 : "memory");
}
__device__ __forceinline__ void wait_async3() {
#if __has_builtin(__builtin_amdgcn_s_wait_asynccnt)
    __builtin_amdgcn_s_wait_asynccnt(3);
#else
    asm volatile("s_wait_asynccnt 0x3" ::: "memory");
#endif
}
__device__ __forceinline__ void wait_async0() {
#if __has_builtin(__builtin_amdgcn_s_wait_asynccnt)
    __builtin_amdgcn_s_wait_asynccnt(0);
#else
    asm volatile("s_wait_asynccnt 0x0" ::: "memory");
#endif
}

// --- WMMA helpers ----------------------------------------------------------
__device__ __forceinline__ v16bf ld_frag_lds(const char* p) {
    v8bf lo = *(const v8bf*)(p);        // 16B
    v8bf hi = *(const v8bf*)(p + 16);   // 16B
    return __builtin_shufflevector(lo, hi, 0, 1, 2, 3, 4, 5, 6, 7,
                                           8, 9, 10, 11, 12, 13, 14, 15);
}
__device__ __forceinline__ v8f wmma_bf16(v16bf a, v16bf b, v8f c) {
    return __builtin_amdgcn_wmma_f32_16x16x32_bf16(
        false, a, false, b, (short)0, c, false, false);
}
__device__ __forceinline__ float sigmoidf_(float x) {
    return 1.0f / (1.0f + __expf(-x));
}

// LDS layout per buffer (12 KB):
//   A region: 8 tiles (one per wave) * 32 lanes * 32B   = 8192 B
//   B region: 4 gate tiles          * 32 lanes * 32B    = 4096 B
// Fragment order: lane L's 32B at [tile*1024 + L*32]:
//   A: bytes 0..15 = A[m0+(L&15), k+8*(L>>4) .. +7], 16..31 = same row, +16 cols
//   B: bytes 0..15 = W[n0+(L&15), k+8*(L>>4) .. +7], 16..31 = same row, +16 cols
#define BUF_BYTES 12288
#define B_REGION  8192

// One pipelined GEMM phase: acc[g] += A[m,:]*W[g*HID+n,:]^T over kTot columns.
__device__ __forceinline__ void gemm_phase(
    const __bf16* __restrict__ A, const __bf16* __restrict__ W,
    int ld, int kTot, int mh, int n0t,
    int wave, int lane, int tid, char* smem, v8f acc[4])
{
    // ---- decode this thread's 3 staging chunks (16B each, 768 total) ----
    unsigned int ldsoff[3];
    const __bf16* gp[3];
#pragma unroll
    for (int c = 0; c < 3; ++c) {
        int cid = tid + c * 256;                 // 0..767
        if (cid < 512) {                         // A chunks
            int a = cid >> 6, rem = cid & 63;
            int l = rem >> 1, h = rem & 1;
            ldsoff[c] = (unsigned)(a * 1024 + l * 32 + h * 16);
            int row = mh * 128 + a * 16 + (l & 15);
            int col = ((l >> 4) << 3) + (h << 4);
            gp[c] = A + (long)row * ld + col;
        } else {                                 // B chunks
            int cid2 = cid - 512;
            int g = cid2 >> 6, rem = cid2 & 63;
            int l = rem >> 1, h = rem & 1;
            ldsoff[c] = (unsigned)(B_REGION + g * 1024 + l * 32 + h * 16);
            int row = g * HID + n0t * 16 + (l & 15);
            int col = ((l >> 4) << 3) + (h << 4);
            gp[c] = W + (long)row * ld + col;
        }
    }

    const int nk = kTot >> 5;
    // prologue: fill buffer 0
#pragma unroll
    for (int c = 0; c < 3; ++c)
        async_copy_b128(smem + ldsoff[c], gp[c]);

    for (int i = 0; i < nk; ++i) {
        const int buf = i & 1;
        if (i + 1 < nk) {                        // prefetch next buffer
            const int koff = (i + 1) << 5;
#pragma unroll
            for (int c = 0; c < 3; ++c)
                async_copy_b128(smem + (buf ^ 1) * BUF_BYTES + ldsoff[c],
                                gp[c] + koff);
            wait_async3();                       // current buffer landed
        } else {
            wait_async0();
        }
        __syncthreads();                         // visible to all waves

        const char* base = smem + buf * BUF_BYTES;
        v16bf a = ld_frag_lds(base + wave * 1024 + lane * 32);
#pragma unroll
        for (int g = 0; g < 4; ++g) {
            v16bf b = ld_frag_lds(base + B_REGION + g * 1024 + lane * 32);
            acc[g] = wmma_bf16(a, b, acc[g]);
        }
        __syncthreads();                         // reads done before re-fill
    }
}

// --- utility kernels -------------------------------------------------------
__global__ void k_f32_to_bf16(const float* __restrict__ s,
                              __bf16* __restrict__ d, int n) {
    int i = blockIdx.x * blockDim.x + threadIdx.x;
    int stride = gridDim.x * blockDim.x;
    for (; i < n; i += stride) d[i] = (__bf16)s[i];
}
__global__ void k_zero_f32(float* __restrict__ d, int n) {
    int i = blockIdx.x * blockDim.x + threadIdx.x;
    int stride = gridDim.x * blockDim.x;
    for (; i < n; i += stride) d[i] = 0.0f;
}
__global__ void k_zero_bf16(__bf16* __restrict__ d, int n) {
    int i = blockIdx.x * blockDim.x + threadIdx.x;
    int stride = gridDim.x * blockDim.x;
    for (; i < n; i += stride) d[i] = (__bf16)0.0f;
}

// --- fused LSTM cell -------------------------------------------------------
// grid = 2 batch-halves * 64 n-tiles = 128 blocks of 256 threads (8 waves).
// wave w owns the 16x16 tile (m0 = mh*128 + w*16, n0 = ntile*16) and
// accumulates all four gate tiles so the LSTM pointwise update fuses in.
__global__ void __launch_bounds__(256)
k_lstm_cell(const __bf16* __restrict__ X,    // [BATCH, Kx]
            int Kx,
            const __bf16* __restrict__ Hprev,// [BATCH, HID]
            const __bf16* __restrict__ Wih,  // [4*HID, Kx]
            const __bf16* __restrict__ Whh,  // [4*HID, HID]
            const float* __restrict__ bih,
            const float* __restrict__ bhh,
            float* __restrict__ Cst,         // [BATCH, HID] in/out
            __bf16* __restrict__ Hout)       // [BATCH, HID]
{
    __shared__ char smem[2 * BUF_BYTES];
    const int tid  = threadIdx.x;
    const int lane = tid & 31;
    const int wave = tid >> 5;
    const int n0t  = blockIdx.x & 63;
    const int mh   = blockIdx.x >> 6;

    v8f acc[4];
#pragma unroll
    for (int g = 0; g < 4; ++g)
        acc[g] = (v8f){0.f, 0.f, 0.f, 0.f, 0.f, 0.f, 0.f, 0.f};

    gemm_phase(X,     Wih, Kx,  Kx,  mh, n0t, wave, lane, tid, smem, acc);
    gemm_phase(Hprev, Whh, HID, HID, mh, n0t, wave, lane, tid, smem, acc);

    // ---- fused gate epilogue ----
    const int n = n0t * 16 + (lane & 15);
    const float bi = bih[0 * HID + n] + bhh[0 * HID + n];
    const float bf = bih[1 * HID + n] + bhh[1 * HID + n];
    const float bg = bih[2 * HID + n] + bhh[2 * HID + n];
    const float bo = bih[3 * HID + n] + bhh[3 * HID + n];
    const int mbase = mh * 128 + wave * 16 + ((lane >> 4) << 3);

#pragma unroll
    for (int r = 0; r < 8; ++r) {
        const long idx = (long)(mbase + r) * HID + n;
        float i_ = sigmoidf_(acc[0][r] + bi);
        float f_ = sigmoidf_(acc[1][r] + bf);
        float g_ = tanhf    (acc[2][r] + bg);
        float o_ = sigmoidf_(acc[3][r] + bo);
        float cn = f_ * Cst[idx] + i_ * g_;
        Cst[idx]  = cn;
        Hout[idx] = (__bf16)(o_ * tanhf(cn));
    }
}

// --- output projection: out[t] = h3 @ W_out^T + b_out ----------------------
__global__ void __launch_bounds__(256)
k_out_gemm(const __bf16* __restrict__ H3,
           const __bf16* __restrict__ Wout,
           const float* __restrict__ bout,
           float* __restrict__ out)
{
    const int lane = threadIdx.x & 31;
    const int wave = (blockIdx.x << 3) + (threadIdx.x >> 5);   // 0..255
    const int m0 = (wave >> 4) << 4;
    const int n0 = (wave & 15) << 4;

    v8f acc = (v8f){0.f, 0.f, 0.f, 0.f, 0.f, 0.f, 0.f, 0.f};
    for (int k = 0; k < HID; k += 32) {
        // direct-load fragments (small GEMM; weights stay hot in L2)
        int r  = m0 + (lane & 15);
        int kb = k + ((lane >> 4) << 3);
        const __bf16* pa = H3 + (long)r * HID + kb;
        v8bf alo = *(const v8bf*)(pa);
        v8bf ahi = *(const v8bf*)(pa + 16);
        v16bf a = __builtin_shufflevector(alo, ahi, 0, 1, 2, 3, 4, 5, 6, 7,
                                                    8, 9, 10, 11, 12, 13, 14, 15);
        int nn  = n0 + (lane & 15);
        int kb2 = k + ((lane >> 4) << 3);
        const __bf16* pb = Wout + (long)nn * HID + kb2;
        v8bf blo = *(const v8bf*)(pb);
        v8bf bhi = *(const v8bf*)(pb + 16);
        v16bf b = __builtin_shufflevector(blo, bhi, 0, 1, 2, 3, 4, 5, 6, 7,
                                                    8, 9, 10, 11, 12, 13, 14, 15);
        acc = wmma_bf16(a, b, acc);
    }

    const int n = n0 + (lane & 15);
    const float bb = bout[n];
    const int mbase = m0 + ((lane >> 4) << 3);
#pragma unroll
    for (int r = 0; r < 8; ++r)
        out[(long)(mbase + r) * OUTD + n] = acc[r] + bb;
}

// ---------------------------------------------------------------------------
extern "C" void kernel_launch(void* const* d_in, const int* in_sizes, int n_in,
                              void* d_out, int out_size, void* d_ws, size_t ws_size,
                              hipStream_t stream) {
    const float* input = (const float*)d_in[0];
    const float* W_ih1 = (const float*)d_in[1];
    const float* W_hh1 = (const float*)d_in[2];
    const float* b_ih1 = (const float*)d_in[3];
    const float* b_hh1 = (const float*)d_in[4];
    const float* W_ih2 = (const float*)d_in[5];
    const float* W_hh2 = (const float*)d_in[6];
    const float* b_ih2 = (const float*)d_in[7];
    const float* b_hh2 = (const float*)d_in[8];
    const float* W_ih3 = (const float*)d_in[9];
    const float* W_hh3 = (const float*)d_in[10];
    const float* b_ih3 = (const float*)d_in[11];
    const float* b_hh3 = (const float*)d_in[12];
    const float* W_out = (const float*)d_in[13];
    const float* b_out = (const float*)d_in[14];
    float* out = (float*)d_out;

    // ---- workspace carve-up (256B aligned) ----
    char* p = (char*)d_ws;
    auto alloc = [&](size_t bytes) -> void* {
        void* r = (void*)p;
        p += (bytes + 255) & ~(size_t)255;
        return r;
    };
    __bf16* xb    = (__bf16*)alloc((size_t)SEQ_LEN * BATCH * IN_DIM * 2);
    __bf16* wih1b = (__bf16*)alloc((size_t)4 * HID * IN_DIM * 2);
    __bf16* whh1b = (__bf16*)alloc((size_t)4 * HID * HID * 2);
    __bf16* wih2b = (__bf16*)alloc((size_t)4 * HID * HID * 2);
    __bf16* whh2b = (__bf16*)alloc((size_t)4 * HID * HID * 2);
    __bf16* wih3b = (__bf16*)alloc((size_t)4 * HID * HID * 2);
    __bf16* whh3b = (__bf16*)alloc((size_t)4 * HID * HID * 2);
    __bf16* woutb = (__bf16*)alloc((size_t)OUTD * HID * 2);
    __bf16* h1[2] = { (__bf16*)alloc((size_t)BATCH * HID * 2),
                      (__bf16*)alloc((size_t)BATCH * HID * 2) };
    __bf16* h2[2] = { (__bf16*)alloc((size_t)BATCH * HID * 2),
                      (__bf16*)alloc((size_t)BATCH * HID * 2) };
    __bf16* h3[2] = { (__bf16*)alloc((size_t)BATCH * HID * 2),
                      (__bf16*)alloc((size_t)BATCH * HID * 2) };
    float* c1 = (float*)alloc((size_t)BATCH * HID * 4);
    float* c2 = (float*)alloc((size_t)BATCH * HID * 4);
    float* c3 = (float*)alloc((size_t)BATCH * HID * 4);

    const int CT = 256;
    auto conv = [&](const float* s, __bf16* d, int n) {
        int blocks = (n + CT * 8 - 1) / (CT * 8);
        if (blocks > 1024) blocks = 1024;
        hipLaunchKernelGGL(k_f32_to_bf16, dim3(blocks), dim3(CT), 0, stream, s, d, n);
    };
    conv(input, xb,    SEQ_LEN * BATCH * IN_DIM);
    conv(W_ih1, wih1b, 4 * HID * IN_DIM);
    conv(W_hh1, whh1b, 4 * HID * HID);
    conv(W_ih2, wih2b, 4 * HID * HID);
    conv(W_hh2, whh2b, 4 * HID * HID);
    conv(W_ih3, wih3b, 4 * HID * HID);
    conv(W_hh3, whh3b, 4 * HID * HID);
    conv(W_out, woutb, OUTD * HID);

    hipLaunchKernelGGL(k_zero_f32, dim3(256), dim3(CT), 0, stream, c1, BATCH * HID);
    hipLaunchKernelGGL(k_zero_f32, dim3(256), dim3(CT), 0, stream, c2, BATCH * HID);
    hipLaunchKernelGGL(k_zero_f32, dim3(256), dim3(CT), 0, stream, c3, BATCH * HID);
    hipLaunchKernelGGL(k_zero_bf16, dim3(256), dim3(CT), 0, stream, h1[0], BATCH * HID);
    hipLaunchKernelGGL(k_zero_bf16, dim3(256), dim3(CT), 0, stream, h2[0], BATCH * HID);
    hipLaunchKernelGGL(k_zero_bf16, dim3(256), dim3(CT), 0, stream, h3[0], BATCH * HID);

    int pp = 0;
    for (int t = 0; t < SEQ_LEN; ++t) {
        const __bf16* xt = xb + (size_t)t * BATCH * IN_DIM;

        hipLaunchKernelGGL(k_lstm_cell, dim3(128), dim3(256), 0, stream,
                           xt, IN_DIM, h1[pp], wih1b, whh1b, b_ih1, b_hh1,
                           c1, h1[pp ^ 1]);
        hipLaunchKernelGGL(k_lstm_cell, dim3(128), dim3(256), 0, stream,
                           h1[pp ^ 1], HID, h2[pp], wih2b, whh2b, b_ih2, b_hh2,
                           c2, h2[pp ^ 1]);
        hipLaunchKernelGGL(k_lstm_cell, dim3(128), dim3(256), 0, stream,
                           h2[pp ^ 1], HID, h3[pp], wih3b, whh3b, b_ih3, b_hh3,
                           c3, h3[pp ^ 1]);
        hipLaunchKernelGGL(k_out_gemm, dim3(32), dim3(256), 0, stream,
                           h3[pp ^ 1], woutb, b_out,
                           out + (size_t)t * BATCH * OUTD);
        pp ^= 1;
    }
}
